// PAM_52003464020623
// MI455X (gfx1250) — matrix operand order repaired and checked by reading, not compiled, not run on hardware
//
#include <hip/hip_runtime.h>
#include <hip/hip_bf16.h>
#include <stdint.h>

// PAM attention for MI455X (gfx1250): bf16 WMMA flash-attention pipeline
// with async global->LDS staging of K chunks (ASYNCcnt path).

typedef __attribute__((ext_vector_type(16))) __bf16 v16bf;
typedef __attribute__((ext_vector_type(8)))  __bf16 v8bf;
typedef __attribute__((ext_vector_type(8)))  float  v8f;

#define BB   4
#define CC   512
#define NN   4096
#define MIDD 64
#define JC   128   // keys per chunk (8 waves x 16)

static __device__ __forceinline__ v16bf cat8(v8bf lo, v8bf hi) {
  return __builtin_shufflevector(lo, hi, 0,1,2,3,4,5,6,7,8,9,10,11,12,13,14,15);
}
static __device__ __forceinline__ v8f wmma_bf16(v16bf a, v16bf b, v8f c) {
  // D = A(16x32) * B(32x16) + C, f32 accumulate
  return __builtin_amdgcn_wmma_f32_16x16x32_bf16(false, a, false, b, (short)0, c, false, false);
}

// async 16B copy: global -> LDS, tracked by ASYNCcnt
static __device__ __forceinline__ void async_copy_b128(uint32_t lds_byte_off,
                                                       const void* gptr) {
  asm volatile("global_load_async_to_lds_b128 %0, %1, off"
               :: "v"(lds_byte_off), "v"((unsigned long long)(uintptr_t)gptr)
               : "memory");
}
static __device__ __forceinline__ void wait_asynccnt0() {
  asm volatile("s_wait_asynccnt 0x0" ::: "memory");
}

// ---------------------------------------------------------------------------
// Kernel 0: convert fp32 weights to bf16 in workspace
// ---------------------------------------------------------------------------
__global__ void cvt_weights(const float* __restrict__ Wq, const float* __restrict__ Wk,
                            const float* __restrict__ Wv,
                            __bf16* __restrict__ Wqb, __bf16* __restrict__ Wkb,
                            __bf16* __restrict__ Wvb) {
  int i = blockIdx.x * blockDim.x + threadIdx.x;
  if (i < MIDD * CC) { Wqb[i] = (__bf16)Wq[i]; Wkb[i] = (__bf16)Wk[i]; }
  if (i < CC * CC)   { Wvb[i] = (__bf16)Wv[i]; }
}

// ---------------------------------------------------------------------------
// Kernel 1: projections via WMMA.
//   Q[b][n][m], K[b][n][m] (m=0..63, contiguous m)   -- bf16
//   V[b][d][n] (d=0..511, contiguous n)              -- bf16
// Each wave = one job: 16 rows of n x 64 output cols, K-loop over C=512.
// job 0 -> Q, job 1 -> K, jobs 2..9 -> V channel slab (job-2)*64.
// ---------------------------------------------------------------------------
__global__ __launch_bounds__(256) void proj_kernel(
    const float* __restrict__ x,
    const __bf16* __restrict__ Wqb, const __bf16* __restrict__ Wkb,
    const __bf16* __restrict__ Wvb,
    const float* __restrict__ bq, const float* __restrict__ bk,
    const float* __restrict__ bv,
    __bf16* __restrict__ Qb, __bf16* __restrict__ Kb, __bf16* __restrict__ Vd) {
  const int lane  = threadIdx.x & 31;
  const int wid   = (blockIdx.x * blockDim.x + threadIdx.x) >> 5;
  const int job   = wid % 10;
  const int ntile = (wid / 10) & 255;
  const int b     = wid / 2560;
  const int n0    = ntile * 16;
  const int lrow  = lane & 15;   // A row / B,D column within tile
  const int lhi   = lane >> 4;   // half-wave select

  const float* xb = x + (size_t)b * CC * NN;

  const __bf16* W; const float* bias; int mbase;
  if (job == 0)      { W = Wqb; bias = bq; mbase = 0; }
  else if (job == 1) { W = Wkb; bias = bk; mbase = 0; }
  else               { W = Wvb; bias = bv; mbase = (job - 2) * 64; }

  v8f acc[4] = {};

  for (int c0 = 0; c0 < CC; c0 += 32) {
    // A fragment: x^T tile rows n0..n0+15, K = c0..c0+31 (16-bit A layout)
    v16bf a;
#pragma unroll
    for (int i2 = 0; i2 < 16; ++i2) {
      int c = c0 + lhi * 16 + i2;
      a[i2] = (__bf16)xb[(size_t)c * NN + n0 + lrow];
    }
#pragma unroll
    for (int dt = 0; dt < 4; ++dt) {
      int m = mbase + dt * 16 + lrow;          // B column
      const __bf16* wp = W + (size_t)m * CC + c0 + lhi * 16;  // K contiguous
      v16bf bfr = cat8(*(const v8bf*)wp, *(const v8bf*)(wp + 8));
      acc[dt] = wmma_bf16(a, bfr, acc[dt]);
    }
  }

  if (job <= 1) {
    __bf16* outp = ((job == 0) ? Qb : Kb) + (size_t)b * NN * MIDD;
#pragma unroll
    for (int dt = 0; dt < 4; ++dt) {
      int m = dt * 16 + lrow;
      float bi = bias[m];
#pragma unroll
      for (int r = 0; r < 8; ++r) {
        int n = n0 + lhi * 8 + r;              // D layout: VGPR r -> row
        outp[(size_t)n * MIDD + m] = (__bf16)(acc[dt][r] + bi);
      }
    }
  } else {
    __bf16* outp = Vd + (size_t)b * CC * NN;
#pragma unroll
    for (int dt = 0; dt < 4; ++dt) {
      int d = mbase + dt * 16 + lrow;
      float bi = bias[d];
      v8bf pk;
#pragma unroll
      for (int r = 0; r < 8; ++r) pk[r] = (__bf16)(acc[dt][r] + bi);
      // 8 consecutive n per lane -> one 16B store
      *(v8bf*)(outp + (size_t)d * NN + n0 + lhi * 8) = pk;
    }
  }
}

// ---------------------------------------------------------------------------
// Kernel 2: flash attention + residual.
// One workgroup (8 waves) per (b, 16-query tile). Chunk over 128 keys.
//   K chunk (16KB) is staged into LDS via global_load_async_to_lds_b128,
//   double-buffered one chunk ahead (ASYNCcnt).
//   wave w: S subtile for keys j0+16w (2 WMMAs from LDS K), online softmax
//   via half-wave shfl reductions + LDS cross-wave reduction, P -> LDS bf16,
//   then PV: wave w accumulates O[16 x 64] for channel slab w*64 (16 WMMAs).
// ---------------------------------------------------------------------------
__global__ __launch_bounds__(256) void attn_kernel(
    const __bf16* __restrict__ Qb, const __bf16* __restrict__ Kb,
    const __bf16* __restrict__ Vd, const float* __restrict__ x,
    const float* __restrict__ gamma, float* __restrict__ out) {
  __shared__ __align__(16) __bf16 K_lds[2][JC][MIDD];   // 32 KB double buffer
  __shared__ __align__(16) __bf16 P_lds[16][JC];        // 4 KB
  __shared__ float wave_max[8][16];
  __shared__ float wave_sum[8][16];
  __shared__ float row_m[16];
  __shared__ float row_l[16];

  const int tid  = threadIdx.x;
  const int lane = tid & 31;
  const int w    = tid >> 5;
  const int lrow = lane & 15;
  const int lhi  = lane >> 4;
  const int b    = blockIdx.x >> 8;
  const int i0   = (blockIdx.x & 255) * 16;

  const __bf16* Qp = Qb + (size_t)b * NN * MIDD;
  const __bf16* Kp = Kb + (size_t)b * NN * MIDD;
  const __bf16* Vp = Vd + (size_t)b * CC * NN;

  if (tid < 16) { row_m[tid] = -3.0e38f; row_l[tid] = 0.0f; }

  // Q A-fragments for K = 0..31 and 32..63 (held for the whole kernel)
  v16bf aq0, aq1;
  {
    const __bf16* qrow = Qp + (size_t)(i0 + lrow) * MIDD;
    aq0 = cat8(*(const v8bf*)(qrow + lhi * 16),      *(const v8bf*)(qrow + 8 + lhi * 16));
    aq1 = cat8(*(const v8bf*)(qrow + 32 + lhi * 16), *(const v8bf*)(qrow + 40 + lhi * 16));
  }

  // prologue: stage K chunk 0 into buffer 0 (each thread copies 4 x 16B;
  // chunk is a contiguous JC*MIDD*2 = 16KB block of Kb)
  {
    const uint32_t lds0 = (uint32_t)(uintptr_t)&K_lds[0][0][0];
#pragma unroll
    for (int q = 0; q < 4; ++q) {
      const int p = tid * 4 + q;                 // 0..1023 16B pieces
      async_copy_b128(lds0 + (uint32_t)p * 16, Kp + (size_t)p * 8);
    }
  }

  v8f o[4] = {};

  for (int j0 = 0; j0 < NN; j0 += JC) {
    const int buf = (j0 / JC) & 1;
    const int jw  = j0 + w * 16;

    // wait for this wave's staged copies, then make them visible to all waves
    wait_asynccnt0();
    __syncthreads();  // chunk-start barrier: K_lds[buf] ready; prior reads of buf^1 done

    // stage next chunk into the other buffer (overlaps with compute below)
    if (j0 + JC < NN) {
      const uint32_t lds1 = (uint32_t)(uintptr_t)&K_lds[buf ^ 1][0][0];
      const __bf16* gsrc = Kp + (size_t)(j0 + JC) * MIDD;
#pragma unroll
      for (int q = 0; q < 4; ++q) {
        const int p = tid * 4 + q;
        async_copy_b128(lds1 + (uint32_t)p * 16, gsrc + (size_t)p * 8);
      }
      // prefetch next V lines for this wave's channel slab
      __builtin_prefetch(Vp + (size_t)(w * 64 + lrow) * NN + j0 + JC, 0, 1);
    }

    // ---- S = Q * K^T subtile (16 queries x 16 keys), K-dim = 64, K from LDS ----
    v8f s = {};
    {
      const __bf16* krow = &K_lds[buf][w * 16 + lrow][lhi * 16];
      v16bf bk0 = cat8(*(const v8bf*)(krow),      *(const v8bf*)(krow + 8));
      v16bf bk1 = cat8(*(const v8bf*)(krow + 32), *(const v8bf*)(krow + 40));
      s = wmma_bf16(aq0, bk0, s);
      s = wmma_bf16(aq1, bk1, s);
    }

    // ---- per-row partial max within this wave's 16 keys ----
    float pm[8];
#pragma unroll
    for (int r = 0; r < 8; ++r) {
      float v = s[r];
      v = fmaxf(v, __shfl_xor(v, 1, 32));
      v = fmaxf(v, __shfl_xor(v, 2, 32));
      v = fmaxf(v, __shfl_xor(v, 4, 32));
      v = fmaxf(v, __shfl_xor(v, 8, 32));
      pm[r] = v;
    }
    if (lrow == 0) {
#pragma unroll
      for (int r = 0; r < 8; ++r) wave_max[w][lhi * 8 + r] = pm[r];
    }
    __syncthreads();  // barrier 1: wave_max visible

    // ---- online softmax update ----
    float nm[8], al[8], ps[8];
#pragma unroll
    for (int r = 0; r < 8; ++r) {
      const int row = lhi * 8 + r;
      float mc = wave_max[0][row];
#pragma unroll
      for (int ww = 1; ww < 8; ++ww) mc = fmaxf(mc, wave_max[ww][row]);
      const float old  = row_m[row];
      const float newm = fmaxf(old, mc);
      nm[r] = newm;
      al[r] = __expf(old - newm);
      const float p = __expf(s[r] - newm);
      P_lds[row][w * 16 + lrow] = (__bf16)p;
      float t = p;
      t += __shfl_xor(t, 1, 32);
      t += __shfl_xor(t, 2, 32);
      t += __shfl_xor(t, 4, 32);
      t += __shfl_xor(t, 8, 32);
      ps[r] = t;
      o[0][r] *= al[r]; o[1][r] *= al[r]; o[2][r] *= al[r]; o[3][r] *= al[r];
    }
    if (lrow == 0) {
#pragma unroll
      for (int r = 0; r < 8; ++r) wave_sum[w][lhi * 8 + r] = ps[r];
    }
    __syncthreads();  // barrier 2: P_lds + wave_sum visible

    if (w == 0 && lrow == 0) {
#pragma unroll
      for (int r = 0; r < 8; ++r) {
        const int row = lhi * 8 + r;
        float tot = 0.0f;
#pragma unroll
        for (int ww = 0; ww < 8; ++ww) tot += wave_sum[ww][row];
        row_l[row] = row_l[row] * al[r] + tot;
        row_m[row] = nm[r];
      }
    }

    // ---- O += P * V (wave w owns channels [w*64, w*64+64)) ----
#pragma unroll
    for (int kk = 0; kk < 4; ++kk) {
      const __bf16* prow = &P_lds[lrow][kk * 32 + lhi * 16];
      v16bf pA = cat8(*(const v8bf*)(prow), *(const v8bf*)(prow + 8));
#pragma unroll
      for (int dt = 0; dt < 4; ++dt) {
        const int d = w * 64 + dt * 16 + lrow;
        const __bf16* vrow = Vp + (size_t)d * NN + j0 + kk * 32 + lhi * 16;
        v16bf vB = cat8(*(const v8bf*)(vrow), *(const v8bf*)(vrow + 8));
        o[dt] = wmma_bf16(pA, vB, o[dt]);
      }
    }
    // next chunk's P_lds / wave_max / K_lds writes are fenced by the
    // chunk-start barrier + barrier 1 above
  }
  __syncthreads();

  float linv[8];
#pragma unroll
  for (int r = 0; r < 8; ++r) linv[r] = 1.0f / row_l[lhi * 8 + r];
  const float g = gamma[0];
  const float* xb = x   + (size_t)b * CC * NN;
  float*       ob = out + (size_t)b * CC * NN;
#pragma unroll
  for (int dt = 0; dt < 4; ++dt) {
    const int d = w * 64 + dt * 16 + lrow;
#pragma unroll
    for (int r = 0; r < 8; ++r) {
      const int n = i0 + lhi * 8 + r;
      const size_t idx = (size_t)d * NN + n;
      ob[idx] = g * (o[dt][r] * linv[r]) + xb[idx];
    }
  }
}

// ---------------------------------------------------------------------------
extern "C" void kernel_launch(void* const* d_in, const int* in_sizes, int n_in,
                              void* d_out, int out_size, void* d_ws, size_t ws_size,
                              hipStream_t stream) {
  const float* x     = (const float*)d_in[0];
  const float* Wq    = (const float*)d_in[1];
  const float* bq    = (const float*)d_in[2];
  const float* Wk    = (const float*)d_in[3];
  const float* bk    = (const float*)d_in[4];
  const float* Wv    = (const float*)d_in[5];
  const float* bv    = (const float*)d_in[6];
  const float* gamma = (const float*)d_in[7];
  float* out = (float*)d_out;

  char* ws = (char*)d_ws;
  __bf16* Wqb = (__bf16*)ws;  ws += (size_t)MIDD * CC * 2;
  __bf16* Wkb = (__bf16*)ws;  ws += (size_t)MIDD * CC * 2;
  __bf16* Wvb = (__bf16*)ws;  ws += (size_t)CC * CC * 2;
  __bf16* Qb  = (__bf16*)ws;  ws += (size_t)BB * NN * MIDD * 2;
  __bf16* Kbf = (__bf16*)ws;  ws += (size_t)BB * NN * MIDD * 2;
  __bf16* Vd  = (__bf16*)ws;  ws += (size_t)BB * CC * NN * 2;

  cvt_weights<<<(CC * CC + 255) / 256, 256, 0, stream>>>(Wq, Wk, Wv, Wqb, Wkb, Wvb);

  // B * 256 n-tiles * 10 wave-jobs, 8 waves per block
  proj_kernel<<<(BB * 256 * 10) / 8, 256, 0, stream>>>(
      x, Wqb, Wkb, Wvb, bq, bk, bv, Qb, Kbf, Vd);

  // one block per (b, 16-query tile)
  attn_kernel<<<BB * (NN / 16), 256, 0, stream>>>(Qb, Kbf, Vd, x, gamma, out);
}
